// PointTransformerSegment_661424963761
// MI455X (gfx1250) — matrix operand order repaired and checked by reading, not compiled, not run on hardware
//
#include <hip/hip_runtime.h>
#include <hip/hip_bf16.h>

typedef __attribute__((ext_vector_type(16))) __bf16    v16bf;
typedef __attribute__((ext_vector_type(8)))  float     v8f;
typedef __attribute__((ext_vector_type(8)))  unsigned  v8u;

#define BB 4
#define NN 4096
#define DD 64
#define KK 16
#define BN (BB*NN)
#define PAD 68   // LDS row pitch in floats for staged tiles (16B-aligned rows)

// A/B fragments carried as 8 packed dwords; bit-cast at the WMMA call.
#define WMMA_BF16(a,b,c)                                                   \
  __builtin_amdgcn_wmma_f32_16x16x32_bf16(false,                           \
      __builtin_bit_cast(v16bf,(a)), false,                                \
      __builtin_bit_cast(v16bf,(b)), (short)0, (c), false, false)

// Hardware pack-convert: two f32 -> one dword of two bf16 (lo = a, hi = b).
static __device__ __forceinline__ unsigned pk2u(float a, float b) {
  unsigned r;
  asm("v_cvt_pk_bf16_f32 %0, %1, %2" : "=v"(r) : "v"(a), "v"(b));
  return r;
}

static __device__ __forceinline__ __bf16 f2bf(float f) {
  unsigned u = pk2u(f, 0.f);
  unsigned short h = (unsigned short)u;
  return __builtin_bit_cast(__bf16, h);
}

// A-fragment element->K map (ISA 7.12.2, 16-bit A 16x32): two contiguous runs:
// elements 0..7  -> K = base + e        (base = lane<16 ? 0 : 8)
// elements 8..15 -> K = base + 16 + e-8
static __device__ __forceinline__ int a_base(int lane, int ks) {
  return 32 * ks + ((lane < 16) ? 0 : 8);
}

static __device__ __forceinline__ void pack8(v8u& r, int o4, float4 a, float4 b) {
  r[o4 + 0] = pk2u(a.x, a.y);
  r[o4 + 1] = pk2u(a.z, a.w);
  r[o4 + 2] = pk2u(b.x, b.y);
  r[o4 + 3] = pk2u(b.z, b.w);
}

// Build 16x32 bf16 A-fragment from a row-major f32 row (4x 16B loads).
static __device__ __forceinline__ v8u load_a_frag_f32(const float* row, int base) {
  v8u r;
  const float4 a0 = *(const float4*)(row + base);
  const float4 a1 = *(const float4*)(row + base + 4);
  const float4 b0 = *(const float4*)(row + base + 16);
  const float4 b1 = *(const float4*)(row + base + 20);
  pack8(r, 0, a0, a1);
  pack8(r, 4, b0, b1);
  return r;
}

// B-fragment (32x16) from TRANSPOSED bf16 weights in LDS: Wt[n*64 + k].
// lane L: column n = t*16 + L%16; K runs = k0 + (L>=16?8:0) + e and +16 -> two runs of 8 bf16.
static __device__ __forceinline__ v8u load_b_frag_t(const __bf16* Wt, int k0, int t, int lane) {
  const int n  = t * 16 + (lane & 15);
  const int kb = k0 + ((lane >> 4) << 3);
  const uint4 lo = *(const uint4*)(Wt + n * DD + kb);        // 16B, 16B-aligned
  const uint4 hi = *(const uint4*)(Wt + n * DD + kb + 16);   // 16B, 16B-aligned
  v8u r;
  r[0] = lo.x; r[1] = lo.y; r[2] = lo.z; r[3] = lo.w;
  r[4] = hi.x; r[5] = hi.y; r[6] = hi.z; r[7] = hi.w;
  return r;
}

// ---------------------------------------------------------------- kNN -------
__global__ __launch_bounds__(256) void knn_kernel(const float* __restrict__ pos,
                                                  int* __restrict__ idx_out) {
  const int b = blockIdx.x >> 4;                       // 16 blocks per batch
  const int n = ((blockIdx.x & 15) << 8) + threadIdx.x;
  __shared__ float sp[256][3];

  const float px = pos[(size_t)(b * NN + n) * 3 + 0];
  const float py = pos[(size_t)(b * NN + n) * 3 + 1];
  const float pz = pos[(size_t)(b * NN + n) * 3 + 2];

  float dk[KK]; int ik[KK];
#pragma unroll
  for (int k = 0; k < KK; ++k) { dk[k] = 3.4e38f; ik[k] = 0; }

  for (int tile = 0; tile < NN / 256; ++tile) {
    const int j0 = tile << 8;
    __syncthreads();
    sp[threadIdx.x][0] = pos[(size_t)(b * NN + j0 + threadIdx.x) * 3 + 0];
    sp[threadIdx.x][1] = pos[(size_t)(b * NN + j0 + threadIdx.x) * 3 + 1];
    sp[threadIdx.x][2] = pos[(size_t)(b * NN + j0 + threadIdx.x) * 3 + 2];
    __syncthreads();
    for (int j = 0; j < 256; ++j) {
      const float dx = px - sp[j][0], dy = py - sp[j][1], dz = pz - sp[j][2];
      const float d = dx * dx + dy * dy + dz * dz;
      if (d < dk[KK - 1]) {                         // sorted insertion (1 bubble pass)
        dk[KK - 1] = d; ik[KK - 1] = j0 + j;
#pragma unroll
        for (int t = KK - 2; t >= 0; --t) {
          if (dk[t + 1] < dk[t]) {
            float td = dk[t]; dk[t] = dk[t + 1]; dk[t + 1] = td;
            int   ti = ik[t]; ik[t] = ik[t + 1]; ik[t + 1] = ti;
          }
        }
      }
    }
  }
#pragma unroll
  for (int k = 0; k < KK; ++k) idx_out[(size_t)(b * NN + n) * KK + k] = ik[k];
}

// ------------------------------------------- q/k/v projections (WMMA) -------
__global__ __launch_bounds__(128) void proj_kernel(const float* __restrict__ x,
                                                   const float* __restrict__ Wq,
                                                   const float* __restrict__ Wk,
                                                   const float* __restrict__ Wv,
                                                   float* __restrict__ qf,
                                                   float* __restrict__ kf,
                                                   float* __restrict__ vf) {
  __shared__ __bf16 sW[3][DD * DD];                 // transposed: [n*64 + k]
  const int tid = threadIdx.x;
  for (int i = tid; i < DD * DD; i += 128) {
    const int kk = i >> 6, nn2 = i & 63;            // read contiguous, write transposed
    sW[0][nn2 * DD + kk] = f2bf(Wq[i]);
    sW[1][nn2 * DD + kk] = f2bf(Wk[i]);
    sW[2][nn2 * DD + kk] = f2bf(Wv[i]);
  }
  __syncthreads();

  const int lane = tid & 31, wv = tid >> 5, half = lane >> 4;
  const int rowbase = (blockIdx.x * 4 + wv) * 16;   // 1024 row tiles total
  const float* xr = x + ((size_t)rowbase + (lane & 15)) * DD;

  v8u aA[2];
#pragma unroll
  for (int ks = 0; ks < 2; ++ks) aA[ks] = load_a_frag_f32(xr, a_base(lane, ks));

  float* outs[3] = {qf, kf, vf};
#pragma unroll
  for (int w = 0; w < 3; ++w) {
#pragma unroll
    for (int t = 0; t < 4; ++t) {
      v8f acc = {};
      acc = WMMA_BF16(aA[0], load_b_frag_t(sW[w], 0, t, lane), acc);
      acc = WMMA_BF16(aA[1], load_b_frag_t(sW[w], 32, t, lane), acc);
      const int col = t * 16 + (lane & 15);
#pragma unroll
      for (int v = 0; v < 8; ++v)
        outs[w][(size_t)(rowbase + v + half * 8) * DD + col] = acc[v];
    }
  }
}

// ------------------- fused per-point attention (24 WMMAs / point) ----------
__global__ __launch_bounds__(64) void fused_kernel(const float* __restrict__ pos,
                                                   const int* __restrict__ nn_idx,
                                                   const float* __restrict__ qf,
                                                   const float* __restrict__ kf,
                                                   const float* __restrict__ vf,
                                                   const float* __restrict__ P1,
                                                   const float* __restrict__ p1b,
                                                   const float* __restrict__ P2,
                                                   const float* __restrict__ p2b,
                                                   const float* __restrict__ A1,
                                                   const float* __restrict__ a1b,
                                                   const float* __restrict__ A2,
                                                   const float* __restrict__ a2b,
                                                   float* __restrict__ agg) {
  __shared__ __bf16 sP2[DD * DD], sA1[DD * DD], sA2[DD * DD];   // transposed
  __shared__ float sP1[3 * DD], sp1b[DD], sp2b[DD], sa1b[DD], sa2b[DD];
  __shared__ float sPE[2][KK][PAD];     // pos_enc staged per wave
  __shared__ float sH2[2][KK][PAD];     // relu(t@A1) staged per wave
  __shared__ int   sIdx[2][KK];

  const int tid = threadIdx.x;
  for (int i = tid; i < DD * DD; i += 64) {
    const int kk = i >> 6, nn2 = i & 63;
    sP2[nn2 * DD + kk] = f2bf(P2[i]);
    sA1[nn2 * DD + kk] = f2bf(A1[i]);
    sA2[nn2 * DD + kk] = f2bf(A2[i]);
  }
  for (int i = tid; i < 3 * DD; i += 64) sP1[i] = P1[i];
  if (tid < DD) { sp1b[tid] = p1b[tid]; sp2b[tid] = p2b[tid];
                  sa1b[tid] = a1b[tid]; sa2b[tid] = a2b[tid]; }
  __syncthreads();

  const int lane = tid & 31, wv = tid >> 5, half = lane >> 4, mrow = lane & 15;

  // Hoist all loop-invariant B fragments into registers (24 x 8 VGPRs).
  v8u bP2[2][4], bA1[2][4], bA2[2][4];
#pragma unroll
  for (int ks = 0; ks < 2; ++ks)
#pragma unroll
    for (int t = 0; t < 4; ++t) {
      bP2[ks][t] = load_b_frag_t(sP2, 32 * ks, t, lane);
      bA1[ks][t] = load_b_frag_t(sA1, 32 * ks, t, lane);
      bA2[ks][t] = load_b_frag_t(sA2, 32 * ks, t, lane);
    }

  const int gw = blockIdx.x * 2 + wv, nw = gridDim.x * 2;

  for (int p = gw; p < BN; p += nw) {            // uniform per wave: 16384/1024
    const int b = p >> 12, n = p & (NN - 1);

    if (lane < KK) sIdx[wv][lane] = nn_idx[(size_t)p * KK + lane];
    asm volatile("s_wait_dscnt 0" ::: "memory");
    const int nb = sIdx[wv][mrow];

    // rel for my row m = mrow
    const float rx = pos[(size_t)(b * NN + n) * 3 + 0] - pos[(size_t)(b * NN + nb) * 3 + 0];
    const float ry = pos[(size_t)(b * NN + n) * 3 + 1] - pos[(size_t)(b * NN + nb) * 3 + 1];
    const float rz = pos[(size_t)(b * NN + n) * 3 + 2] - pos[(size_t)(b * NN + nb) * 3 + 2];

    // hidden = relu(rel @ P1 + p1b), built directly in A-fragment layout
    v8u hA[2];
#pragma unroll
    for (int ks = 0; ks < 2; ++ks) {
      const int base = a_base(lane, ks);
#pragma unroll
      for (int run = 0; run < 2; ++run)
#pragma unroll
        for (int e = 0; e < 4; ++e) {                       // pairs of channels
          const int c = base + 16 * run + 2 * e;
          const float h0 = fmaxf(rx * sP1[c]     + ry * sP1[DD + c]     + rz * sP1[2 * DD + c]     + sp1b[c],     0.f);
          const float h1 = fmaxf(rx * sP1[c + 1] + ry * sP1[DD + c + 1] + rz * sP1[2 * DD + c + 1] + sp1b[c + 1], 0.f);
          hA[ks][run * 4 + e] = pk2u(h0, h1);
        }
    }

    // pos_enc = hidden @ P2 + p2b  -> stage to LDS (C-layout -> row-major)
#pragma unroll
    for (int t = 0; t < 4; ++t) {
      v8f acc = {};
      acc = WMMA_BF16(hA[0], bP2[0][t], acc);
      acc = WMMA_BF16(hA[1], bP2[1][t], acc);
      const int col = t * 16 + (lane & 15);
#pragma unroll
      for (int v = 0; v < 8; ++v)
        sPE[wv][v + half * 8][col] = acc[v] + sp2b[col];
    }
    asm volatile("s_wait_dscnt 0" ::: "memory");

    // t = q - kf + pos_enc, in A-fragment layout (vectorized 16B loads)
    const float* qr = qf + (size_t)p * DD;
    const float* kr = kf + (size_t)(b * NN + nb) * DD;
    const float* pr = &sPE[wv][mrow][0];
    v8u tA[2];
#pragma unroll
    for (int ks = 0; ks < 2; ++ks) {
      const int base = a_base(lane, ks);
#pragma unroll
      for (int run = 0; run < 2; ++run) {
        const int c0 = base + 16 * run;
        const float4 q0 = *(const float4*)(qr + c0),     q1 = *(const float4*)(qr + c0 + 4);
        const float4 k0 = *(const float4*)(kr + c0),     k1 = *(const float4*)(kr + c0 + 4);
        const float4 e0 = *(const float4*)(pr + c0),     e1 = *(const float4*)(pr + c0 + 4);
        float4 u0, u1;
        u0.x = q0.x - k0.x + e0.x; u0.y = q0.y - k0.y + e0.y;
        u0.z = q0.z - k0.z + e0.z; u0.w = q0.w - k0.w + e0.w;
        u1.x = q1.x - k1.x + e1.x; u1.y = q1.y - k1.y + e1.y;
        u1.z = q1.z - k1.z + e1.z; u1.w = q1.w - k1.w + e1.w;
        pack8(tA[ks], run * 4, u0, u1);
      }
    }

    // h2 = relu(t @ A1 + a1b) -> stage to LDS
#pragma unroll
    for (int t = 0; t < 4; ++t) {
      v8f acc = {};
      acc = WMMA_BF16(tA[0], bA1[0][t], acc);
      acc = WMMA_BF16(tA[1], bA1[1][t], acc);
      const int col = t * 16 + (lane & 15);
#pragma unroll
      for (int v = 0; v < 8; ++v)
        sH2[wv][v + half * 8][col] = fmaxf(acc[v] + sa1b[col], 0.f);
    }
    asm volatile("s_wait_dscnt 0" ::: "memory");

    v8u h2A[2];
#pragma unroll
    for (int ks = 0; ks < 2; ++ks)
      h2A[ks] = load_a_frag_f32(&sH2[wv][mrow][0], a_base(lane, ks));

    // attn = softmax_K((h2 @ A2 + a2b)/8); agg = sum_k attn*(v+pos_enc)
#pragma unroll
    for (int t = 0; t < 4; ++t) {
      v8f acc = {};
      acc = WMMA_BF16(h2A[0], bA2[0][t], acc);
      acc = WMMA_BF16(h2A[1], bA2[1][t], acc);
      const int col = t * 16 + (lane & 15);
      const float bias = sa2b[col];
      float lg[8]; float mx = -3.4e38f;
#pragma unroll
      for (int v = 0; v < 8; ++v) { lg[v] = (acc[v] + bias) * 0.125f; mx = fmaxf(mx, lg[v]); }
      mx = fmaxf(mx, __shfl_xor(mx, 16, 32));      // other 8 rows live in lane^16
      float s = 0.f;
#pragma unroll
      for (int v = 0; v < 8; ++v) { lg[v] = __expf(lg[v] - mx); s += lg[v]; }
      s += __shfl_xor(s, 16, 32);
      const float inv = 1.f / s;
      float part = 0.f;
#pragma unroll
      for (int v = 0; v < 8; ++v) {
        const int m = v + half * 8;
        const int nbm = sIdx[wv][m];
        part += lg[v] * inv * (vf[(size_t)(b * NN + nbm) * DD + col] + sPE[wv][m][col]);
      }
      part += __shfl_xor(part, 16, 32);
      if (half == 0) agg[(size_t)p * DD + col] = part;
    }
  }
}

// ------------------------ out = agg @ Wf + bf + x (WMMA + residual) --------
__global__ __launch_bounds__(128) void final_kernel(const float* __restrict__ agg,
                                                    const float* __restrict__ Wf,
                                                    const float* __restrict__ bf,
                                                    const float* __restrict__ x,
                                                    float* __restrict__ out) {
  __shared__ __bf16 sW[DD * DD];                    // transposed
  __shared__ float  sb[DD];
  const int tid = threadIdx.x;
  for (int i = tid; i < DD * DD; i += 128) {
    const int kk = i >> 6, nn2 = i & 63;
    sW[nn2 * DD + kk] = f2bf(Wf[i]);
  }
  if (tid < DD) sb[tid] = bf[tid];
  __syncthreads();

  const int lane = tid & 31, wv = tid >> 5, half = lane >> 4;
  const int rowbase = (blockIdx.x * 4 + wv) * 16;
  const float* ar = agg + ((size_t)rowbase + (lane & 15)) * DD;

  v8u aA[2];
#pragma unroll
  for (int ks = 0; ks < 2; ++ks) aA[ks] = load_a_frag_f32(ar, a_base(lane, ks));

#pragma unroll
  for (int t = 0; t < 4; ++t) {
    v8f acc = {};
    acc = WMMA_BF16(aA[0], load_b_frag_t(sW, 0, t, lane), acc);
    acc = WMMA_BF16(aA[1], load_b_frag_t(sW, 32, t, lane), acc);
    const int col = t * 16 + (lane & 15);
#pragma unroll
    for (int v = 0; v < 8; ++v) {
      const size_t r = (size_t)(rowbase + v + half * 8);
      out[r * DD + col] = acc[v] + sb[col] + x[r * DD + col];
    }
  }
}

// ---------------------------------------------------------------------------
extern "C" void kernel_launch(void* const* d_in, const int* in_sizes, int n_in,
                              void* d_out, int out_size, void* d_ws, size_t ws_size,
                              hipStream_t stream) {
  const float* x   = (const float*)d_in[0];
  const float* pos = (const float*)d_in[1];
  const float* Wq  = (const float*)d_in[2];
  const float* Wk  = (const float*)d_in[3];
  const float* Wv  = (const float*)d_in[4];
  const float* P1  = (const float*)d_in[5];
  const float* p1b = (const float*)d_in[6];
  const float* P2  = (const float*)d_in[7];
  const float* p2b = (const float*)d_in[8];
  const float* A1  = (const float*)d_in[9];
  const float* a1b = (const float*)d_in[10];
  const float* A2  = (const float*)d_in[11];
  const float* a2b = (const float*)d_in[12];
  const float* Wf  = (const float*)d_in[13];
  const float* bf  = (const float*)d_in[14];
  float* out = (float*)d_out;

  char* ws = (char*)d_ws;
  int*   idx = (int*)  (ws);                               // 1 MB
  float* qf  = (float*)(ws + (size_t)(1  << 20));          // 4 MB
  float* kf  = (float*)(ws + (size_t)(5  << 20));          // 4 MB
  float* vf  = (float*)(ws + (size_t)(9  << 20));          // 4 MB
  float* agg = (float*)(ws + (size_t)(13 << 20));          // 4 MB

  knn_kernel  <<<BB * (NN / 256), 256, 0, stream>>>(pos, idx);
  proj_kernel <<<BN / (16 * 4), 128, 0, stream>>>(x, Wq, Wk, Wv, qf, kf, vf);
  fused_kernel<<<512, 64, 0, stream>>>(pos, idx, qf, kf, vf,
                                       P1, p1b, P2, p2b, A1, a1b, A2, a2b, agg);
  final_kernel<<<BN / (16 * 4), 128, 0, stream>>>(agg, Wf, bf, x, out);
}
